// FineMatching_22986664969105
// MI455X (gfx1250) — matrix operand order repaired and checked by reading, not compiled, not run on hardware
//
#include <hip/hip_runtime.h>
#include <math.h>

typedef _Float16 half_t;
typedef __attribute__((ext_vector_type(16))) _Float16 v16h;
typedef __attribute__((ext_vector_type(4)))  _Float16 h4;
typedef __attribute__((ext_vector_type(8)))  float    v8f;
typedef __attribute__((ext_vector_type(4)))  unsigned int v4u;
typedef int v4i __attribute__((vector_size(16)));   // matches builtin's param type

#define BB 2
#define CCH 128
#define HH 128
#define WWD 128
#define HW (HH*WWD)
#define PIX (BB*HW)
#define NITERS 10

// CDNA5 async global->LDS staging (ASYNCcnt path), guarded so absence of the
// builtin on a given toolchain falls back to plain b128 copies.
#if defined(__HIP_DEVICE_COMPILE__) && \
    __has_builtin(__builtin_amdgcn_global_load_async_to_lds_b128) && \
    __has_builtin(__builtin_amdgcn_s_wait_asynccnt)
#define USE_ASYNC_LDS 1
#else
#define USE_ASYNC_LDS 0
#endif

__device__ __forceinline__ float gelu_f(float x){ return 0.5f*x*(1.0f + erff(x*0.70710678118654752f)); }
// 16-bit B-matrix 32x16 K map (table order: k = 8h+s for s<8, 16+8h+(s-8) for s>=8)
__device__ __forceinline__ int kmapB(int j,int kh){ return kh*8+2*j+(j>=4?8:0); }

// ---------------- packing ----------------
__global__ void pack_chlast(const float* __restrict__ in, half_t* __restrict__ out){
  int pix = blockIdx.x*blockDim.x + threadIdx.x; if(pix>=PIX) return;
  int b = pix / HW, p = pix % HW;
  for(int c=0;c<CCH;++c)
    out[(size_t)pix*CCH + c] = (half_t)in[((size_t)b*CCH + c)*HW + p];
}

// OIHW (or OI) f32 -> B-fragment-ready f16: [tap][kc][NT_total slot nt_base+nt][lane*16+e]
__global__ void prepack_w(const float* __restrict__ w, half_t* __restrict__ wpk,
                          int Cin_real, int KC, int NT_total, int nt_base, int NT_src,
                          int KH, int total){
  int idx = blockIdx.x*blockDim.x + threadIdx.x; if(idx>=total) return;
  int e = idx & 15, l = (idx>>4)&31, blk = idx>>9;
  int nt = blk % NT_src, t2 = blk / NT_src, kc = t2 % KC, tap = t2 / KC;
  int j = e>>1, hb = e&1, kh = l>>4;
  int K = kmapB(j,kh) + hb;
  int i = kc*32 + K;
  int o = nt*16 + (l&15);
  float v = 0.f;
  if (i < Cin_real){
    int ky = tap / KH, kx = tap % KH;
    v = w[((size_t)(o*Cin_real + i)*KH + ky)*KH + kx];
  }
  size_t widx = ((size_t)((tap*KC+kc)*NT_total + nt_base + nt))*512 + l*16 + e;
  wpk[widx] = (half_t)v;
}

__global__ void concat_bias(const float* __restrict__ a, const float* __restrict__ b,
                            float* __restrict__ dst){
  int i = threadIdx.x;
  dst[i] = (i < 128) ? a[i] : b[i-128];
}

// ---------------- WMMA implicit-GEMM conv (3x3 or 1x1), channel-last f16 ----------------
// Fully templated so the TAPS*KC K-loop unrolls into straight-line WMMA code.
template<int TAPS, int KC, int NT, int ACT>
__global__ void conv_wmma(const half_t* __restrict__ in, const half_t* __restrict__ wpk,
                          const float* __restrict__ bias, half_t* __restrict__ out,
                          int Cout){
  constexpr int Cin  = KC*32;
  constexpr int ROWS = (TAPS==9)?3:1;
  constexpr int PXW  = (TAPS==9)?18:16;
  constexpr int ROWH = PXW*Cin;          // halves per halo row
  constexpr int CH16 = ROWH/8;           // 16B chunks per halo row
  extern __shared__ half_t s_in[];

  int tile = blockIdx.x;
  int tpb = HW/16;
  int b  = tile / tpb; int rp = tile % tpb;
  int p0 = rp*16; int y = p0 / WWD; int x0 = p0 % WWD;
  int lane = threadIdx.x & 31; int wv = threadIdx.x >> 5;
  int yb = (TAPS==9)? y-1 : y; int xb = (TAPS==9)? x0-1 : x0;

  // ---- stage zero-padded halo tile into LDS ----
  bool interior = (TAPS==1) || (y>=1 && y<=HH-2 && x0>=1 && x0+17<=WWD);
  if (interior){
    // halo = ROWS contiguous channel-last segments
#if USE_ASYNC_LDS
    for(int i=threadIdx.x; i<ROWS*CH16; i+=NT*32){
      int r = i / CH16, o = i % CH16;
      const half_t* src = in + ((size_t)(b*HH + yb + r)*WWD + xb)*Cin + o*8;
      half_t* dst = s_in + r*ROWH + o*8;
      __builtin_amdgcn_global_load_async_to_lds_b128(
          (__attribute__((address_space(1))) v4i*)src,
          (__attribute__((address_space(3))) v4i*)dst, 0, 0);
    }
#else
    for(int i=threadIdx.x; i<ROWS*CH16; i+=NT*32){
      int r = i / CH16, o = i % CH16;
      const v4u* src = (const v4u*)(in + ((size_t)(b*HH + yb + r)*WWD + xb)*Cin);
      ((v4u*)(s_in + r*ROWH))[o] = src[o];
    }
#endif
  } else {
    unsigned int* s32 = (unsigned int*)s_in;
    for(int i=threadIdx.x; i<ROWS*ROWH/2; i+=NT*32){
      int h2 = i*2; int c = h2 % Cin; int pr = h2 / Cin;
      int px = pr % PXW; int r = pr / PXW;
      int yy = yb + r, xx = xb + px;
      unsigned int val = 0u;
      if (yy>=0 && yy<HH && xx>=0 && xx<WWD)
        val = *(const unsigned int*)(in + ((size_t)(b*HH+yy)*WWD + xx)*Cin + c);
      s32[i] = val;
    }
  }
#if USE_ASYNC_LDS
  __builtin_amdgcn_s_wait_asynccnt(0);
#endif
  __syncthreads();

  int m = lane & 15, kh = lane >> 4;
  float bval = bias[wv*16 + m];
  const v16h* wbase = (const v16h*)wpk;

  v8f acc = {};
  #pragma unroll
  for(int tap=0; tap<TAPS; ++tap){
    int tr = (TAPS==9)? tap/3 : 0;
    int tc = (TAPS==9)? tap%3 : 0;
    int rowbase = (tr*PXW + (m+tc))*Cin;
    #pragma unroll
    for(int kc=0; kc<KC; ++kc){
      // A-fragment: explicit 16-bit A layout = two contiguous 16B groups per lane
      const half_t* rb = s_in + rowbase + kc*32;
      union { v4u u4[2]; v16h h; } a;
      a.u4[0] = *(const v4u*)(rb + kh*8);
      a.u4[1] = *(const v4u*)(rb + 16 + kh*8);
      // B-fragment: prepacked, per-lane contiguous 32B
      v16h bf = wbase[(size_t)((tap*KC+kc)*NT + wv)*32 + lane];
      acc = __builtin_amdgcn_wmma_f32_16x16x32_f16(false, a.h, false, bf,
                                                   (short)0, acc, false, false);
    }
  }

  #pragma unroll
  for(int r=0;r<8;++r){
    int M = r + (kh?8:0);
    float v = acc[r] + bval;
    if (ACT) v = gelu_f(v);
    out[((size_t)b*HW + p0 + M)*Cout + wv*16 + m] = (half_t)v;
  }
}

// ---------------- warp(lf1,flow), diff, L2-normalize, ad = 1 - diff (wave/pixel) ----------------
__global__ void warp_diff(const half_t* __restrict__ lf0, const half_t* __restrict__ lf1,
                          const float* __restrict__ flow, half_t* __restrict__ ad){
  int gw = (blockIdx.x*blockDim.x + threadIdx.x) >> 5;
  int lane = threadIdx.x & 31;
  if (gw >= PIX) return;
  int b = gw / HW, p = gw % HW, y = p / WWD, x = p % WWD;
  float fx = flow[(size_t)gw*2+0], fy = flow[(size_t)gw*2+1];
  float xx = fminf(fmaxf((float)x + fx, 0.f), (float)(WWD-1));
  float yy = fminf(fmaxf((float)y + fy, 0.f), (float)(HH-1));
  float x0f = floorf(xx), y0f = floorf(yy);
  int x0 = (int)x0f, y0 = (int)y0f;
  int x1 = min(x0+1, WWD-1), y1 = min(y0+1, HH-1);
  float wx = xx - x0f, wy = yy - y0f;
  float w00=(1.f-wx)*(1.f-wy), w01=wx*(1.f-wy), w10=(1.f-wx)*wy, w11=wx*wy;
  size_t base = (size_t)b*HW;
  int c0 = lane*4;
  h4 v00 = *(const h4*)(lf1 + (base + (size_t)y0*WWD + x0)*CCH + c0);
  h4 v01 = *(const h4*)(lf1 + (base + (size_t)y0*WWD + x1)*CCH + c0);
  h4 v10 = *(const h4*)(lf1 + (base + (size_t)y1*WWD + x0)*CCH + c0);
  h4 v11 = *(const h4*)(lf1 + (base + (size_t)y1*WWD + x1)*CCH + c0);
  h4 f0  = *(const h4*)(lf0 + (size_t)gw*CCH + c0);
  float d[4]; float ss = 0.f;
  #pragma unroll
  for(int e=0;e<4;++e){
    float wvv = (float)v00[e]*w00 + (float)v01[e]*w01 + (float)v10[e]*w10 + (float)v11[e]*w11;
    d[e] = (float)f0[e] - wvv; ss += d[e]*d[e];
  }
  #pragma unroll
  for(int off=16;off>0;off>>=1) ss += __shfl_xor(ss, off, 32);
  float rn = 1.0f / fmaxf(sqrtf(ss), 1e-12f);
  h4 o;
  #pragma unroll
  for(int e=0;e<4;++e) o[e] = (half_t)(1.0f - d[e]*rn);
  *(h4*)(ad + (size_t)gw*CCH + c0) = o;
}

// ---------------- 25-tap local attention + softmax (wave/pixel) ----------------
// q/k fused in one channel-last buffer: qk[pix][256] = [q(128) | k(128)]
__global__ void attn(const half_t* __restrict__ qk,
                     const float* __restrict__ flow, float* __restrict__ fi){
  int gw = (blockIdx.x*blockDim.x + threadIdx.x) >> 5;
  int lane = threadIdx.x & 31;
  if (gw >= PIX) return;
  int b = gw/HW, p = gw%HW, y = p/WWD, x = p%WWD;
  int c0 = lane*4;
  h4 qv = *(const h4*)(qk + (size_t)gw*256 + c0);
  float q0=qv[0],q1=qv[1],q2=qv[2],q3=qv[3];
  const float inv = 0.08838834764831845f; // 1/sqrt(128)
  float sc[25];
  #pragma unroll
  for(int t=0;t<25;++t){
    int ny = y + t/5 - 2, nx = x + t%5 - 2;
    float s = 0.f;
    if (ny>=0 && ny<HH && nx>=0 && nx<WWD){   // zero-padded unfold: OOB score = 0
      size_t np = (size_t)b*HW + (size_t)ny*WWD + nx;
      h4 kv = *(const h4*)(qk + np*256 + 128 + c0);
      s = q0*(float)kv[0]+q1*(float)kv[1]+q2*(float)kv[2]+q3*(float)kv[3];
    }
    #pragma unroll
    for(int off=16;off>0;off>>=1) s += __shfl_xor(s, off, 32);
    sc[t] = s*inv;
  }
  float mx = sc[0];
  #pragma unroll
  for(int t=1;t<25;++t) mx = fmaxf(mx, sc[t]);
  float se = 0.f;
  #pragma unroll
  for(int t=0;t<25;++t){ sc[t] = expf(sc[t]-mx); se += sc[t]; }
  float rs = 1.f/se;
  if (lane==0){
    float ax=0.f, ay=0.f;
    #pragma unroll
    for(int t=0;t<25;++t){
      int ny = y + t/5 - 2, nx = x + t%5 - 2;
      if (ny>=0 && ny<HH && nx>=0 && nx<WWD){ // zero-padded flow window
        size_t np = (size_t)b*HW + (size_t)ny*WWD + nx;
        ax += sc[t]*rs*flow[np*2+0];
        ay += sc[t]*rs*flow[np*2+1];
      }
    }
    fi[(size_t)gw*2+0]=ax; fi[(size_t)gw*2+1]=ay;
  }
}

// ---------------- tiny refine net (thread/pixel) ----------------
__global__ void fr_conv1(const float* __restrict__ fi, const float* __restrict__ w,
                         const float* __restrict__ bias, float* __restrict__ t1){
  int pix = blockIdx.x*blockDim.x + threadIdx.x; if(pix>=PIX) return;
  int b=pix/HW,p=pix%HW,y=p/WWD,x=p%WWD;
  float acc[16];
  #pragma unroll
  for(int o=0;o<16;++o) acc[o]=bias[o];
  for(int ky=0;ky<3;++ky){ int ny=y+ky-1; if(ny<0||ny>=HH) continue;
    for(int kx=0;kx<3;++kx){ int nx=x+kx-1; if(nx<0||nx>=WWD) continue;
      size_t np=(size_t)b*HW+(size_t)ny*WWD+nx;
      float f0=fi[np*2], f1=fi[np*2+1];
      #pragma unroll
      for(int o=0;o<16;++o)
        acc[o] += w[((o*2+0)*3+ky)*3+kx]*f0 + w[((o*2+1)*3+ky)*3+kx]*f1;
  }}
  #pragma unroll
  for(int o=0;o<16;++o) t1[(size_t)pix*16+o]=gelu_f(acc[o]);
}

__global__ void fr_conv2(const float* __restrict__ t1, const float* __restrict__ w,
                         const float* __restrict__ bias, float* __restrict__ t2){
  int pix = blockIdx.x*blockDim.x + threadIdx.x; if(pix>=PIX) return;
  int b=pix/HW,p=pix%HW,y=p/WWD,x=p%WWD;
  float acc[16];
  #pragma unroll
  for(int o=0;o<16;++o) acc[o]=bias[o];
  for(int ky=0;ky<3;++ky){ int ny=y+ky-1; if(ny<0||ny>=HH) continue;
    for(int kx=0;kx<3;++kx){ int nx=x+kx-1; if(nx<0||nx>=WWD) continue;
      size_t np=(size_t)b*HW+(size_t)ny*WWD+nx;
      #pragma unroll
      for(int i=0;i<16;++i){
        float v=t1[np*16+i];
        #pragma unroll
        for(int o=0;o<16;++o) acc[o]+=w[((o*16+i)*3+ky)*3+kx]*v;
      }
  }}
  #pragma unroll
  for(int o=0;o<16;++o) t2[(size_t)pix*16+o]=gelu_f(acc[o]);
}

__global__ void fr_conv3_add(const float* __restrict__ t2, const float* __restrict__ w,
                             const float* __restrict__ bias, float* __restrict__ flow){
  int pix = blockIdx.x*blockDim.x + threadIdx.x; if(pix>=PIX) return;
  int b=pix/HW,p=pix%HW,y=p/WWD,x=p%WWD;
  float a0=bias[0], a1=bias[1];
  for(int ky=0;ky<3;++ky){ int ny=y+ky-1; if(ny<0||ny>=HH) continue;
    for(int kx=0;kx<3;++kx){ int nx=x+kx-1; if(nx<0||nx>=WWD) continue;
      size_t np=(size_t)b*HW+(size_t)ny*WWD+nx;
      #pragma unroll
      for(int i=0;i<16;++i){
        float v=t2[np*16+i];
        a0 += w[((0*16+i)*3+ky)*3+kx]*v;
        a1 += w[((1*16+i)*3+ky)*3+kx]*v;
      }
  }}
  flow[(size_t)pix*2+0]+=a0; flow[(size_t)pix*2+1]+=a1;
}

// ---------------- hcat = [feat0 | warp(feat1,flow) | flow | zero-pad] 288ch (wave/pixel) ----------------
__global__ void build_hcat(const half_t* __restrict__ f0h, const half_t* __restrict__ f1h,
                           const float* __restrict__ flow, half_t* __restrict__ hcat){
  int gw = (blockIdx.x*blockDim.x + threadIdx.x) >> 5;
  int lane = threadIdx.x & 31;
  if (gw >= PIX) return;
  int b=gw/HW,p=gw%HW,y=p/WWD,x=p%WWD;
  float fx = flow[(size_t)gw*2+0], fy = flow[(size_t)gw*2+1];
  float xx = fminf(fmaxf((float)x + fx, 0.f), (float)(WWD-1));
  float yy = fminf(fmaxf((float)y + fy, 0.f), (float)(HH-1));
  float x0f = floorf(xx), y0f = floorf(yy);
  int x0=(int)x0f, y0=(int)y0f;
  int x1=min(x0+1,WWD-1), y1=min(y0+1,HH-1);
  float wx=xx-x0f, wy=yy-y0f;
  float w00=(1.f-wx)*(1.f-wy), w01=wx*(1.f-wy), w10=(1.f-wx)*wy, w11=wx*wy;
  size_t base=(size_t)b*HW;
  int c0=lane*4;
  h4 a = *(const h4*)(f0h + (size_t)gw*CCH + c0);
  *(h4*)(hcat + (size_t)gw*288 + c0) = a;
  h4 v00 = *(const h4*)(f1h + (base + (size_t)y0*WWD + x0)*CCH + c0);
  h4 v01 = *(const h4*)(f1h + (base + (size_t)y0*WWD + x1)*CCH + c0);
  h4 v10 = *(const h4*)(f1h + (base + (size_t)y1*WWD + x0)*CCH + c0);
  h4 v11 = *(const h4*)(f1h + (base + (size_t)y1*WWD + x1)*CCH + c0);
  h4 wv;
  #pragma unroll
  for(int e=0;e<4;++e)
    wv[e] = (half_t)((float)v00[e]*w00 + (float)v01[e]*w01 + (float)v10[e]*w10 + (float)v11[e]*w11);
  *(h4*)(hcat + (size_t)gw*288 + 128 + c0) = wv;
  if (lane < 8){
    int bc = 256 + lane*4;
    h4 z;
    #pragma unroll
    for(int e=0;e<4;++e){
      int c = bc+e;
      float v = (c==256)?fx:((c==257)?fy:0.f);
      z[e] = (half_t)v;
    }
    *(h4*)(hcat + (size_t)gw*288 + bc) = z;
  }
}

// ---------------- confidence head 64->1 + sigmoid ----------------
__global__ void cf3_sigmoid(const half_t* __restrict__ h2, const float* __restrict__ w,
                            const float* __restrict__ bias, float* __restrict__ out){
  int pix = blockIdx.x*blockDim.x + threadIdx.x; if(pix>=PIX) return;
  int b=pix/HW,p=pix%HW,y=p/WWD,x=p%WWD;
  float acc=bias[0];
  for(int ky=0;ky<3;++ky){ int ny=y+ky-1; if(ny<0||ny>=HH) continue;
    for(int kx=0;kx<3;++kx){ int nx=x+kx-1; if(nx<0||nx>=WWD) continue;
      size_t np=(size_t)b*HW+(size_t)ny*WWD+nx;
      #pragma unroll
      for(int i=0;i<64;++i)
        acc += w[(i*3+ky)*3+kx]*(float)h2[np*64+i];
  }}
  out[pix] = 1.f/(1.f+expf(-acc));
}

// ---------------- flow layout conversions ----------------
__global__ void flow_in(const float* __restrict__ src, float* __restrict__ flowc){
  int pix = blockIdx.x*blockDim.x + threadIdx.x; if(pix>=PIX) return;
  int b=pix/HW,p=pix%HW;
  flowc[(size_t)pix*2+0]=src[((size_t)b*2+0)*HW+p];
  flowc[(size_t)pix*2+1]=src[((size_t)b*2+1)*HW+p];
}
__global__ void flow_out(const float* __restrict__ flowc, float* __restrict__ dst){
  int pix = blockIdx.x*blockDim.x + threadIdx.x; if(pix>=PIX) return;
  int b=pix/HW,p=pix%HW;
  dst[((size_t)b*2+0)*HW+p]=flowc[(size_t)pix*2+0];
  dst[((size_t)b*2+1)*HW+p]=flowc[(size_t)pix*2+1];
}

// ---------------- host ----------------
extern "C" void kernel_launch(void* const* d_in, const int* in_sizes, int n_in,
                              void* d_out, int out_size, void* d_ws, size_t ws_size,
                              hipStream_t stream) {
  const float* feat0   = (const float*)d_in[0];
  const float* feat1   = (const float*)d_in[1];
  const float* flow0   = (const float*)d_in[2];
  const float* lc_w1   = (const float*)d_in[3];
  const float* lc_b1   = (const float*)d_in[4];
  const float* lc_w2   = (const float*)d_in[5];
  const float* lc_b2   = (const float*)d_in[6];
  const float* qw      = (const float*)d_in[7];
  const float* qbias   = (const float*)d_in[8];
  const float* kw      = (const float*)d_in[9];
  const float* kbias   = (const float*)d_in[10];
  const float* fr_w1   = (const float*)d_in[11];
  const float* fr_b1   = (const float*)d_in[12];
  const float* fr_w2   = (const float*)d_in[13];
  const float* fr_b2   = (const float*)d_in[14];
  const float* fr_w3   = (const float*)d_in[15];
  const float* fr_b3   = (const float*)d_in[16];
  const float* cf_w1   = (const float*)d_in[17];
  const float* cf_b1   = (const float*)d_in[18];
  const float* cf_w2   = (const float*)d_in[19];
  const float* cf_b2   = (const float*)d_in[20];
  const float* cf_w3   = (const float*)d_in[21];
  const float* cf_b3   = (const float*)d_in[22];

  char* base = (char*)d_ws;
  size_t off = 0;
  auto carve = [&](size_t bytes)->char*{
    size_t cur = (off + 255) & ~(size_t)255;
    off = cur + bytes;
    return base + cur;
  };
  const size_t F128 = (size_t)PIX*CCH*sizeof(half_t);
  half_t* feat0h  = (half_t*)carve(F128);
  half_t* feat1h  = (half_t*)carve(F128);
  half_t* lf0     = (half_t*)carve(F128);
  half_t* lf1     = (half_t*)carve(F128);
  half_t* scratch = (half_t*)carve(F128);           // lc-mid / ad / h1 (non-overlapping lifetimes)
  half_t* qk      = (half_t*)carve((size_t)PIX*256*sizeof(half_t)); // fused q|k
  half_t* hcat    = (half_t*)carve((size_t)PIX*288*sizeof(half_t));
  half_t* h2      = (half_t*)carve((size_t)PIX*64*sizeof(half_t));
  float*  fi      = (float*)carve((size_t)PIX*2*sizeof(float));
  float*  t1      = (float*)carve((size_t)PIX*16*sizeof(float));
  float*  t2      = (float*)carve((size_t)PIX*16*sizeof(float));
  float*  flowc   = (float*)carve((size_t)PIX*2*sizeof(float));
  float*  qkbias  = (float*)carve(256*sizeof(float));
  const int TLC = 9*4*8*512;   // taps*KC*NT_src*512
  const int TQK = 1*4*8*512;   // per q/k half
  const int TC1 = 9*9*8*512;
  const int TC2 = 9*4*4*512;
  half_t* lc1p = (half_t*)carve((size_t)TLC*sizeof(half_t));
  half_t* lc2p = (half_t*)carve((size_t)TLC*sizeof(half_t));
  half_t* qkwp = (half_t*)carve((size_t)(1*4*16*512)*sizeof(half_t)); // NT_total=16
  half_t* cf1p = (half_t*)carve((size_t)TC1*sizeof(half_t));
  half_t* cf2p = (half_t*)carve((size_t)TC2*sizeof(half_t));

  const int tiles = PIX/16;                 // 2048 M-tiles
  const dim3 bpix(PIX/256), tpix(256);      // thread-per-pixel kernels
  const dim3 bwav(PIX*32/256), twav(256);   // wave-per-pixel kernels

  pack_chlast<<<bpix, tpix, 0, stream>>>(feat0, feat0h);
  pack_chlast<<<bpix, tpix, 0, stream>>>(feat1, feat1h);

  prepack_w<<<(TLC+255)/256, 256, 0, stream>>>(lc_w1, lc1p, 128, 4, 8, 0, 8, 3, TLC);
  prepack_w<<<(TLC+255)/256, 256, 0, stream>>>(lc_w2, lc2p, 128, 4, 8, 0, 8, 3, TLC);
  prepack_w<<<(TQK+255)/256, 256, 0, stream>>>(qw,    qkwp, 128, 4, 16, 0, 8, 1, TQK);
  prepack_w<<<(TQK+255)/256, 256, 0, stream>>>(kw,    qkwp, 128, 4, 16, 8, 8, 1, TQK);
  prepack_w<<<(TC1+255)/256, 256, 0, stream>>>(cf_w1, cf1p, 258, 9, 8, 0, 8, 3, TC1);
  prepack_w<<<(TC2+255)/256, 256, 0, stream>>>(cf_w2, cf2p, 128, 4, 4, 0, 4, 3, TC2);
  concat_bias<<<1, 256, 0, stream>>>(qbias, kbias, qkbias);

  const size_t lds33_128 = 3*18*128*sizeof(half_t);  // 13824 B
  const size_t lds11_128 = 1*16*128*sizeof(half_t);  //  4096 B
  const size_t lds33_288 = 3*18*288*sizeof(half_t);  // 31104 B

  // local_conv on both features (hoisted)
  conv_wmma<9,4,8,1><<<tiles, 256, lds33_128, stream>>>(feat0h, lc1p, lc_b1, scratch, 128);
  conv_wmma<9,4,8,0><<<tiles, 256, lds33_128, stream>>>(scratch, lc2p, lc_b2, lf0,    128);
  conv_wmma<9,4,8,1><<<tiles, 256, lds33_128, stream>>>(feat1h, lc1p, lc_b1, scratch, 128);
  conv_wmma<9,4,8,0><<<tiles, 256, lds33_128, stream>>>(scratch, lc2p, lc_b2, lf1,    128);

  flow_in<<<bpix, tpix, 0, stream>>>(flow0, flowc);

  for (int it = 0; it < NITERS; ++it) {
    warp_diff<<<bwav, twav, 0, stream>>>(lf0, lf1, flowc, scratch);            // ad = 1 - diff
    // fused q|k 1x1 GEMM: reads ad once, 16 waves/block, Cout=256
    conv_wmma<1,4,16,0><<<tiles, 512, lds11_128, stream>>>(scratch, qkwp, qkbias, qk, 256);
    attn<<<bwav, twav, 0, stream>>>(qk, flowc, fi);
    fr_conv1<<<bpix, tpix, 0, stream>>>(fi, fr_w1, fr_b1, t1);
    fr_conv2<<<bpix, tpix, 0, stream>>>(t1, fr_w2, fr_b2, t2);
    fr_conv3_add<<<bpix, tpix, 0, stream>>>(t2, fr_w3, fr_b3, flowc);
  }

  build_hcat<<<bwav, twav, 0, stream>>>(feat0h, feat1h, flowc, hcat);
  conv_wmma<9,9,8,1><<<tiles, 256, lds33_288, stream>>>(hcat,    cf1p, cf_b1, scratch, 128); // h1
  conv_wmma<9,4,4,1><<<tiles, 128, lds33_128, stream>>>(scratch, cf2p, cf_b2, h2,      64);
  cf3_sigmoid<<<bpix, tpix, 0, stream>>>(h2, cf_w3, cf_b3, (float*)d_out + (size_t)PIX*2);
  flow_out<<<bpix, tpix, 0, stream>>>(flowc, (float*)d_out);
}
